// QuantLinear_53068615909971
// MI455X (gfx1250) — compile-verified, matrix operand order checked by reading, not run on hardware
//
#include <hip/hip_runtime.h>

// ---------------------------------------------------------------------------
// NF4 dequant + bf16 WMMA GEMM for gfx1250 (MI455X)
//   C[M,N] = X[M,K] * W[N,K]^T,  W = bf16( NF4[q] * absmax[block] )
//   M = 8192 (B*S), N = 16384, K = 4096, absmax block = 64 along K
// ---------------------------------------------------------------------------

typedef __attribute__((ext_vector_type(16))) __bf16 v16bf;
typedef __attribute__((ext_vector_type(8)))  float  v8f;
typedef __attribute__((ext_vector_type(2)))  __bf16 bf16x2;

#define M_DIM 8192
#define N_DIM 16384
#define K_DIM 4096
#define BM 128
#define BN 128
#define BK 64
#define LDSS 72   // bf16 elements per LDS row (64 data + 8 pad -> 144B row)

struct Frag32B { uint4 lo, hi; };

__device__ __forceinline__ unsigned pack_bf16x2(float a, float b) {
    bf16x2 v;
    v.x = (__bf16)a;
    v.y = (__bf16)b;
    return __builtin_bit_cast(unsigned, v);
}

// Load a 16x16(xK32) bf16 fragment for one lane from an LDS tile.
// Lane holds row (passed in), K chunk = kk + (lane>>4)*8 .. +8 and the
// matching +16 chunk (matches the documented 16-bit A/B VGPR layout).
__device__ __forceinline__ v16bf load_frag(const unsigned short* base,
                                           int row, int kk, int lane) {
    const int hi8 = (lane >> 4) * 8;
    const int off = row * LDSS + kk + hi8;
    Frag32B f;
    f.lo = *(const uint4*)(base + off);
    f.hi = *(const uint4*)(base + off + 16);
    return __builtin_bit_cast(v16bf, f);
}

__global__ __launch_bounds__(256)
void nf4_wmma_gemm(const float* __restrict__ x,
                   const int*   __restrict__ q,
                   const float* __restrict__ absmax,
                   float*       __restrict__ out) {
    __shared__ __align__(16) unsigned short sA[BM * LDSS];  // x tile, bf16
    __shared__ __align__(16) unsigned short sB[BN * LDSS];  // W tile, bf16

    const int tid  = threadIdx.x;
    const int lane = tid & 31;
    const int wave = tid >> 5;
    const int n0   = blockIdx.x * BN;
    const int m0   = blockIdx.y * BM;

    // --- NF4 codebook resident in the wave: lane i holds NF4[i & 15] ---
    const int l4 = lane & 15;
    const float tabv =
        (l4 == 0)  ? -1.0f                  :
        (l4 == 1)  ? -0.6961928009986877f   :
        (l4 == 2)  ? -0.5250730514526367f   :
        (l4 == 3)  ? -0.39491748809814453f  :
        (l4 == 4)  ? -0.28444138169288635f  :
        (l4 == 5)  ? -0.18477343022823334f  :
        (l4 == 6)  ? -0.09105003625154495f  :
        (l4 == 7)  ?  0.0f                  :
        (l4 == 8)  ?  0.07958029955625534f  :
        (l4 == 9)  ?  0.16093020141124725f  :
        (l4 == 10) ?  0.24611230194568634f  :
        (l4 == 11) ?  0.33791524171829224f  :
        (l4 == 12) ?  0.44070982933044434f  :
        (l4 == 13) ?  0.5626170039176941f   :
        (l4 == 14) ?  0.7229568362236023f   :  1.0f;
    const int tab_int = __builtin_bit_cast(int, tabv);

    // --- loader mapping: thread covers 32 contiguous K elems of one row ---
    const int lrow = tid >> 1;          // 0..127
    const int lcol = (tid & 1) * 32;    // 0 or 32
    const float* xrow  = x + (size_t)(m0 + lrow) * K_DIM + lcol;
    const int*   qrow  = q + (size_t)(n0 + lrow) * K_DIM + lcol;
    const float* amrow = absmax + (size_t)(n0 + lrow) * (K_DIM / BK);
    const int dsoff = lrow * LDSS + lcol;

    // --- wave tiling: 64(M) x 32(N) per wave; 2x4 wave grid over 128x128 ---
    const int wm = (wave & 1) * 64;
    const int wn = (wave >> 1) * 32;

    v8f acc[4][2];
#pragma unroll
    for (int i = 0; i < 4; ++i)
#pragma unroll
        for (int j = 0; j < 2; ++j)
            acc[i][j] = (v8f){0.f, 0.f, 0.f, 0.f, 0.f, 0.f, 0.f, 0.f};

    for (int ks = 0; ks < K_DIM / BK; ++ks) {
        const float* xp = xrow + ks * BK;
        const int*   qp = qrow + ks * BK;

        // ---- X: 128x64 f32 -> bf16 into LDS ----
#pragma unroll
        for (int v = 0; v < 4; ++v) {
            float4 f0 = ((const float4*)xp)[v * 2 + 0];
            float4 f1 = ((const float4*)xp)[v * 2 + 1];
            uint4 p;
            p.x = pack_bf16x2(f0.x, f0.y);
            p.y = pack_bf16x2(f0.z, f0.w);
            p.z = pack_bf16x2(f1.x, f1.y);
            p.w = pack_bf16x2(f1.z, f1.w);
            *(uint4*)&sA[dsoff + v * 8] = p;
        }

        // ---- W: NF4 codes -> dequant bf16 into LDS ----
        const float am = amrow[ks];  // one absmax per (row, 64-wide K block)
#pragma unroll
        for (int v = 0; v < 4; ++v) {
            int4 qa = ((const int4*)qp)[v * 2 + 0];
            int4 qb = ((const int4*)qp)[v * 2 + 1];
            float c0 = __builtin_bit_cast(float, __builtin_amdgcn_ds_bpermute(qa.x << 2, tab_int)) * am;
            float c1 = __builtin_bit_cast(float, __builtin_amdgcn_ds_bpermute(qa.y << 2, tab_int)) * am;
            float c2 = __builtin_bit_cast(float, __builtin_amdgcn_ds_bpermute(qa.z << 2, tab_int)) * am;
            float c3 = __builtin_bit_cast(float, __builtin_amdgcn_ds_bpermute(qa.w << 2, tab_int)) * am;
            float c4 = __builtin_bit_cast(float, __builtin_amdgcn_ds_bpermute(qb.x << 2, tab_int)) * am;
            float c5 = __builtin_bit_cast(float, __builtin_amdgcn_ds_bpermute(qb.y << 2, tab_int)) * am;
            float c6 = __builtin_bit_cast(float, __builtin_amdgcn_ds_bpermute(qb.z << 2, tab_int)) * am;
            float c7 = __builtin_bit_cast(float, __builtin_amdgcn_ds_bpermute(qb.w << 2, tab_int)) * am;
            uint4 p;
            p.x = pack_bf16x2(c0, c1);
            p.y = pack_bf16x2(c2, c3);
            p.z = pack_bf16x2(c4, c5);
            p.w = pack_bf16x2(c6, c7);
            *(uint4*)&sB[dsoff + v * 8] = p;
        }

        // ---- prefetch next K-step while this tile is consumed ----
        if (ks + 1 < K_DIM / BK) {
            __builtin_prefetch(xp + BK, 0, 1);
            __builtin_prefetch(qp + BK, 0, 1);
        }

        __syncthreads();

        // ---- WMMA over the staged tiles: K = 64 as two K=32 halves ----
#pragma unroll
        for (int kh = 0; kh < 2; ++kh) {
            const int kk = kh * 32;
            v16bf afr[4], bfr[2];
#pragma unroll
            for (int i = 0; i < 4; ++i)
                afr[i] = load_frag(sA, wm + i * 16 + (lane & 15), kk, lane);
#pragma unroll
            for (int j = 0; j < 2; ++j)
                bfr[j] = load_frag(sB, wn + j * 16 + (lane & 15), kk, lane);
#pragma unroll
            for (int i = 0; i < 4; ++i)
#pragma unroll
                for (int j = 0; j < 2; ++j)
                    acc[i][j] = __builtin_amdgcn_wmma_f32_16x16x32_bf16(
                        false, afr[i], false, bfr[j],
                        (short)0, acc[i][j], false, false);
        }

        __syncthreads();
    }

    // ---- epilogue: C layout -> lanes 0-15 N=lane M=r; lanes 16-31 M=r+8 ----
    const int cm = (lane >> 4) * 8;
    const int cn = lane & 15;
#pragma unroll
    for (int i = 0; i < 4; ++i) {
#pragma unroll
        for (int j = 0; j < 2; ++j) {
            size_t base = (size_t)(m0 + wm + i * 16 + cm) * N_DIM
                        + (size_t)(n0 + wn + j * 16 + cn);
#pragma unroll
            for (int r = 0; r < 8; ++r)
                out[base + (size_t)r * N_DIM] = acc[i][j][r];
        }
    }
}

extern "C" void kernel_launch(void* const* d_in, const int* in_sizes, int n_in,
                              void* d_out, int out_size, void* d_ws, size_t ws_size,
                              hipStream_t stream) {
    (void)in_sizes; (void)n_in; (void)out_size; (void)d_ws; (void)ws_size;
    const float* x      = (const float*)d_in[0];
    const int*   q_idx  = (const int*)d_in[1];
    const float* absmax = (const float*)d_in[2];
    float*       out    = (float*)d_out;

    dim3 grid(N_DIM / BN, M_DIM / BM);  // 128 x 64 tiles
    nf4_wmma_gemm<<<grid, 256, 0, stream>>>(x, q_idx, absmax, out);
}
